// RelMultiHeadedSelfAttention_26079041421966
// MI455X (gfx1250) — matrix operand order repaired and checked by reading, not compiled, not run on hardware
//
#include <hip/hip_runtime.h>
#include <hip/hip_bf16.h>

// Problem constants (match reference)
#define B_  4
#define T_  1024
#define D_  512
#define H_  8
#define DK_ 64
#define S_  2047
#define D3_ 1536

typedef __attribute__((ext_vector_type(16))) __bf16        v16bf;
typedef __attribute__((ext_vector_type(8)))  __bf16        v8bf;
typedef __attribute__((ext_vector_type(8)))  float         v8f;
typedef __attribute__((ext_vector_type(4)))  unsigned int  v4u;
typedef __attribute__((ext_vector_type(8)))  int           v8i;
typedef __attribute__((ext_vector_type(4)))  int           v4i;

static __device__ __forceinline__ __bf16 f2bf(float f) {
  union { float f; unsigned u; } x; x.f = f;
  union { unsigned short s; __bf16 b; } y;
  y.s = (unsigned short)(x.u >> 16);
  return y.b;
}

static __device__ __forceinline__ float bf2f(__bf16 b) {
  union { __bf16 b; unsigned short s; } x; x.b = b;
  union { unsigned u; float f; } y;
  y.u = ((unsigned)x.s) << 16;
  return y.f;
}

static __device__ __forceinline__ v16bf cat8(v8bf lo, v8bf hi) {
  return __builtin_shufflevector(lo, hi, 0, 1, 2, 3, 4, 5, 6, 7,
                                 8, 9, 10, 11, 12, 13, 14, 15);
}

// ---------------------------------------------------------------------------
// TDM: DMA one 32-row x 16-col tile (element size via dsz_code: 1=2B, 2=4B)
// from global into LDS, padding each row with 1 DWORD so column reads are
// bank-conflict-free. pad_int chosen so pad fires once per 16-element row.
// D# per CDNA5 ISA sect. 8; 6-arg builtin on this lane.
// ---------------------------------------------------------------------------
static __device__ __forceinline__ void tdm_load_tile_32x16(
    void* lds_dst, const void* gsrc, unsigned row_stride_elems,
    int dsz_code, int pad_int) {
  unsigned long long ga = (unsigned long long)(size_t)gsrc;
  v4u g0;
  g0[0] = 1u;                                                   // count=1
  g0[1] = (unsigned)(size_t)lds_dst;                            // lds_addr (bytes)
  g0[2] = (unsigned)ga;                                         // global_addr[31:0]
  g0[3] = (unsigned)((ga >> 32) & 0x01FFFFFFull) | (2u << 30);  // addr[56:32] | type=2
  v8i g1;
  g1[0] = (int)(((unsigned)dsz_code << 16) | (1u << 20) | ((unsigned)pad_int << 22));
  g1[1] = (int)(16u << 16);                            // tensor_dim0 = 16
  g1[2] = (int)(32u << 16);                            // tensor_dim1 = 32
  g1[3] = (int)(16u << 16);                            // tile_dim0 = 16
  g1[4] = 32;                                          // tile_dim1 = 32 (tile_dim2 = 0)
  g1[5] = (int)row_stride_elems;                       // tensor_dim0_stride[31:0]
  g1[6] = 0;
  g1[7] = 0;
  v4i z4 = {0, 0, 0, 0};
  v8i z8 = {0, 0, 0, 0, 0, 0, 0, 0};
  __builtin_amdgcn_tensor_load_to_lds(g0, g1, z4, z4, z8, 0);
}

#define LDSROW_BF 18  // padded LDS row stride for bf16 tiles (16 + 1 DWORD pad)

// ---------------------------------------------------------------------------
// Elementwise f32 -> bf16 (n must be a multiple of 8)
// ---------------------------------------------------------------------------
__global__ __launch_bounds__(256)
void f32_to_bf16_kernel(const float* __restrict__ src, __bf16* __restrict__ dst,
                        int n8) {
  const int i = blockIdx.x * 256 + threadIdx.x;
  if (i >= n8) return;
  float4 a = ((const float4*)src)[2 * i];
  float4 b = ((const float4*)src)[2 * i + 1];
  v8bf o;
  o[0] = f2bf(a.x); o[1] = f2bf(a.y); o[2] = f2bf(a.z); o[3] = f2bf(a.w);
  o[4] = f2bf(b.x); o[5] = f2bf(b.y); o[6] = f2bf(b.z); o[7] = f2bf(b.w);
  ((v8bf*)dst)[i] = o;
}

// ---------------------------------------------------------------------------
// qu = q + posu, qv = q + posv (bf16 in/out, f32 math).
// One thread per 8 elements of [B*T, D]; posu/posv are flat [H*DK] = [D].
// ---------------------------------------------------------------------------
__global__ __launch_bounds__(256)
void quqv_kernel(const __bf16* __restrict__ qkvb, const float* __restrict__ posu,
                 const float* __restrict__ posv, __bf16* __restrict__ qu,
                 __bf16* __restrict__ qv) {
  const int i   = blockIdx.x * 256 + threadIdx.x;  // over (B*T*D)/8
  const int row = i / (D_ / 8);
  const int g   = i % (D_ / 8);
  const v8bf q = *(const v8bf*)(qkvb + (size_t)row * D3_ + g * 8);
  float4 u0 = ((const float4*)posu)[2 * g], u1 = ((const float4*)posu)[2 * g + 1];
  float4 p0 = ((const float4*)posv)[2 * g], p1 = ((const float4*)posv)[2 * g + 1];
  const float uu[8] = {u0.x, u0.y, u0.z, u0.w, u1.x, u1.y, u1.z, u1.w};
  const float pp[8] = {p0.x, p0.y, p0.z, p0.w, p1.x, p1.y, p1.z, p1.w};
  v8bf ou, ov;
#pragma unroll
  for (int e = 0; e < 8; ++e) {
    const float qf = bf2f(q[e]);
    ou[e] = f2bf(qf + uu[e]);
    ov[e] = f2bf(qf + pp[e]);
  }
  ((v8bf*)qu)[i] = ou;
  ((v8bf*)qv)[i] = ov;
}

// ---------------------------------------------------------------------------
// All-bf16-operand WMMA GEMM: C[M,N] = A[M,K] * W[K,N] (+ f32 bias[N]).
// Block = 4 waves, one 16x16 tile each (shared n0). B tiles staged via TDM
// (bf16, padded rows), double-buffered. OUT_BF16 selects output type.
// ---------------------------------------------------------------------------
template <bool OUT_BF16>
__global__ __launch_bounds__(128)
void wmma_gemm(const __bf16* __restrict__ A, const __bf16* __restrict__ W,
               const float* __restrict__ bias, void* __restrict__ Cout,
               int M, int N, int K) {
  __shared__ __align__(128) __bf16 Bt[2][32 * LDSROW_BF];

  const int n0   = blockIdx.x * 16;
  const int m0   = blockIdx.y * 64 + (threadIdx.x >> 5) * 16;
  const int lane = threadIdx.x & 31;

  int am = m0 + (lane & 15);
  if (am > M - 1) am = M - 1;                 // clamp (store guarded below)
  const int kb  = (lane >> 4) * 8;
  const int bn  = lane & 15;
  const int bkb = (lane >> 4) * 8;

  if (threadIdx.x < 32)                        // wave 0 drives the TDM
    tdm_load_tile_32x16(&Bt[0][0], W + n0, (unsigned)N, 1, 2);

  v8f acc = {};
  int buf = 0;
  for (int k0 = 0; k0 < K; k0 += 32) {
    if (threadIdx.x < 32) {
      if (k0 + 32 < K) {
        tdm_load_tile_32x16(&Bt[buf ^ 1][0], W + (size_t)(k0 + 32) * N + n0,
                            (unsigned)N, 1, 2);
        __builtin_amdgcn_s_wait_tensorcnt(1);
      } else {
        __builtin_amdgcn_s_wait_tensorcnt(0);
      }
    }
    __syncthreads();

    const __bf16* arow = A + (size_t)am * K + k0;
    v16bf a = cat8(*(const v8bf*)(arow + kb), *(const v8bf*)(arow + kb + 16));
    v16bf b;
    const __bf16* bt = &Bt[buf][0];
#pragma unroll
    for (int i = 0; i < 16; ++i) b[i] = bt[(bkb + i + (i >> 3) * 8) * LDSROW_BF + bn];

    acc = __builtin_amdgcn_wmma_f32_16x16x32_bf16(false, a, false, b,
                                                  (short)0, acc, false, false);
    buf ^= 1;
  }

  const int cn  = n0 + (lane & 15);
  const int cmb = m0 + (lane >> 4) * 8;
  const float bv = bias ? bias[cn] : 0.0f;
#pragma unroll
  for (int i = 0; i < 8; ++i) {
    const int cm = cmb + i;
    if (cm < M) {
      if (OUT_BF16) ((__bf16*)Cout)[(size_t)cm * N + cn] = f2bf(acc[i] + bv);
      else          ((float*)Cout)[(size_t)cm * N + cn]  = acc[i] + bv;
    }
  }
}

// ---------------------------------------------------------------------------
// Fused scores: one wave per 16x16 (t,j) tile of one (b,h):
//   AC = qu.k^T (2 WMMAs);  M1 = qv.P_band^T (4 WMMAs, 16x32 p-row band)
//   BD[dt,dj] = M1[dt][15+dj-dt] (Toeplitz gather via LDS)
//   scores = (AC+BD)/8, masked -> f32 weights region of d_out.
// All operands pre-converted bf16: fragment build is pure b128 loads.
// ---------------------------------------------------------------------------
__global__ __launch_bounds__(32)
void scores_kernel(const __bf16* __restrict__ qkvb, const __bf16* __restrict__ qu,
                   const __bf16* __restrict__ qv, const __bf16* __restrict__ pproj,
                   const unsigned char* __restrict__ mask, float* __restrict__ wts) {
  __shared__ float M1s[16][33];

  const int bh = blockIdx.z;
  const int bb = bh / H_;
  const int h  = bh % H_;
  const int t0 = blockIdx.y * 16;
  const int j0 = blockIdx.x * 16;
  const int lane = threadIdx.x;

  const int am  = lane & 15;
  const int kb  = (lane >> 4) * 16;
  const int bn  = lane & 15;
  const int bkb = (lane >> 4) * 16;

  // ---- A fragments (qu / qv rows), K-halves 0..31 and 32..63
  const __bf16* qurow = qu + ((size_t)bb * T_ + (t0 + am)) * D_ + h * DK_;
  const __bf16* qvrow = qv + ((size_t)bb * T_ + (t0 + am)) * D_ + h * DK_;
  v16bf au0 = cat8(*(const v8bf*)(qurow + kb),      *(const v8bf*)(qurow + kb + 8));
  v16bf au1 = cat8(*(const v8bf*)(qurow + 32 + kb), *(const v8bf*)(qurow + 32 + kb + 8));
  v16bf av0 = cat8(*(const v8bf*)(qvrow + kb),      *(const v8bf*)(qvrow + kb + 8));
  v16bf av1 = cat8(*(const v8bf*)(qvrow + 32 + kb), *(const v8bf*)(qvrow + 32 + kb + 8));

  // ---- B fragments for keys: B[dk, n] = k[j0+n, dk] (dk contiguous)
  const __bf16* krow = qkvb + ((size_t)bb * T_ + (j0 + bn)) * D3_ + D_ + h * DK_;
  v16bf bk0 = cat8(*(const v8bf*)(krow + bkb),      *(const v8bf*)(krow + bkb + 8));
  v16bf bk1 = cat8(*(const v8bf*)(krow + 32 + bkb), *(const v8bf*)(krow + 32 + bkb + 8));

  // ---- B fragments for the p band: cols c=0..31 -> p rows r0-15+c
  const int r0 = j0 - t0 + (T_ - 1);
  int s0 = r0 - 15 + (lane & 15);
  int s1 = s0 + 16;
  if (s0 < 0) s0 = 0;  if (s0 > S_ - 1) s0 = S_ - 1;
  if (s1 < 0) s1 = 0;  if (s1 > S_ - 1) s1 = S_ - 1;
  const __bf16* prow0 = pproj + (size_t)s0 * D_ + h * DK_;
  const __bf16* prow1 = pproj + (size_t)s1 * D_ + h * DK_;
  v16bf bp00 = cat8(*(const v8bf*)(prow0 + bkb),      *(const v8bf*)(prow0 + bkb + 8));
  v16bf bp01 = cat8(*(const v8bf*)(prow0 + 32 + bkb), *(const v8bf*)(prow0 + 32 + bkb + 8));
  v16bf bp10 = cat8(*(const v8bf*)(prow1 + bkb),      *(const v8bf*)(prow1 + bkb + 8));
  v16bf bp11 = cat8(*(const v8bf*)(prow1 + 32 + bkb), *(const v8bf*)(prow1 + 32 + bkb + 8));

  // ---- WMMAs
  v8f ac = {};
  ac = __builtin_amdgcn_wmma_f32_16x16x32_bf16(false, au0, false, bk0, (short)0, ac, false, false);
  ac = __builtin_amdgcn_wmma_f32_16x16x32_bf16(false, au1, false, bk1, (short)0, ac, false, false);
  v8f m1a = {}, m1b = {};
  m1a = __builtin_amdgcn_wmma_f32_16x16x32_bf16(false, av0, false, bp00, (short)0, m1a, false, false);
  m1a = __builtin_amdgcn_wmma_f32_16x16x32_bf16(false, av1, false, bp01, (short)0, m1a, false, false);
  m1b = __builtin_amdgcn_wmma_f32_16x16x32_bf16(false, av0, false, bp10, (short)0, m1b, false, false);
  m1b = __builtin_amdgcn_wmma_f32_16x16x32_bf16(false, av1, false, bp11, (short)0, m1b, false, false);

  // ---- spill M1 (16x32) and do the Toeplitz gather
  {
    const int c  = lane & 15;
    const int mb = (lane >> 4) * 8;
#pragma unroll
    for (int i = 0; i < 8; ++i) {
      M1s[mb + i][c]      = m1a[i];
      M1s[mb + i][16 + c] = m1b[i];
    }
  }
  __syncthreads();

  const int dj = lane & 15;
  const int mb = (lane >> 4) * 8;
#pragma unroll
  for (int i = 0; i < 8; ++i) {
    const int dt = mb + i;
    const float bd = M1s[dt][15 + dj - dt];
    const float sc = (ac[i] + bd) * 0.125f;  // 1/sqrt(64)
    const int t = t0 + dt, j = j0 + dj;
    const bool mk = mask[((size_t)bb * T_ + t) * T_ + j] != 0;
    wts[((size_t)bh * T_ + t) * T_ + j] = mk ? sc : -3.402823466e38f;
  }
}

// ---------------------------------------------------------------------------
// Row softmax (T_=1024), in place, b128 I/O. 256 threads per row.
// ---------------------------------------------------------------------------
__global__ __launch_bounds__(256)
void softmax_kernel(float* __restrict__ wts) {
  __shared__ float red[8];
  float* w = wts + (size_t)blockIdx.x * T_;
  const int tid = threadIdx.x;

  float4 vv = ((const float4*)w)[tid];
  float lmax = fmaxf(fmaxf(vv.x, vv.y), fmaxf(vv.z, vv.w));
#pragma unroll
  for (int off = 16; off; off >>= 1) lmax = fmaxf(lmax, __shfl_xor(lmax, off, 32));
  if ((tid & 31) == 0) red[tid >> 5] = lmax;
  __syncthreads();
  float gmax = red[0];
#pragma unroll
  for (int i = 1; i < 8; ++i) gmax = fmaxf(gmax, red[i]);
  __syncthreads();

  vv.x = __expf(vv.x - gmax); vv.y = __expf(vv.y - gmax);
  vv.z = __expf(vv.z - gmax); vv.w = __expf(vv.w - gmax);
  float lsum = (vv.x + vv.y) + (vv.z + vv.w);
#pragma unroll
  for (int off = 16; off; off >>= 1) lsum += __shfl_xor(lsum, off, 32);
  if ((tid & 31) == 0) red[tid >> 5] = lsum;
  __syncthreads();
  float gsum = 0.0f;
#pragma unroll
  for (int i = 0; i < 8; ++i) gsum += red[i];
  const float inv = 1.0f / gsum;
  vv.x *= inv; vv.y *= inv; vv.z *= inv; vv.w *= inv;
  ((float4*)w)[tid] = vv;
}

// ---------------------------------------------------------------------------
// ctx[b,t,h*DK+n] = sum_j W[b,h,t,j] * v[b,j,h*DK+n]   (ctx written bf16)
// Block = 4 waves sharing a TDM-staged bf16 32x16 V tile; A (f32 weights)
// rows loaded b128 + converted (weights must remain f32 in d_out).
// ---------------------------------------------------------------------------
__global__ __launch_bounds__(128)
void context_kernel(const float* __restrict__ wts, const __bf16* __restrict__ qkvb,
                    __bf16* __restrict__ ctx) {
  __shared__ __align__(128) __bf16 Vt[2][32 * LDSROW_BF];

  const int bh = blockIdx.z;
  const int bb = bh / H_;
  const int h  = bh % H_;
  const int t0 = blockIdx.y * 64 + (threadIdx.x >> 5) * 16;
  const int n0 = blockIdx.x * 16;
  const int lane = threadIdx.x & 31;

  const float*  wrow  = wts + (size_t)bh * T_ * T_;
  const __bf16* vbase = qkvb + (size_t)bb * T_ * D3_ + 2 * D_ + h * DK_ + n0;

  const int am  = lane & 15;
  const int kb  = (lane >> 4) * 8;
  const int bn  = lane & 15;
  const int bkb = (lane >> 4) * 8;

  if (threadIdx.x < 32)
    tdm_load_tile_32x16(&Vt[0][0], vbase, (unsigned)D3_, 1, 2);

  v8f acc = {};
  int buf = 0;
  for (int k0 = 0; k0 < T_; k0 += 32) {
    if (threadIdx.x < 32) {
      if (k0 + 32 < T_) {
        tdm_load_tile_32x16(&Vt[buf ^ 1][0], vbase + (size_t)(k0 + 32) * D3_,
                            (unsigned)D3_, 1, 2);
        __builtin_amdgcn_s_wait_tensorcnt(1);
      } else {
        __builtin_amdgcn_s_wait_tensorcnt(0);
      }
    }
    __syncthreads();

    const float* ar = wrow + (size_t)(t0 + am) * T_ + k0;
    float4 a0 = *(const float4*)(ar + kb);
    float4 a1 = *(const float4*)(ar + kb + 4);
    float4 a2 = *(const float4*)(ar + kb + 16);
    float4 a3 = *(const float4*)(ar + kb + 20);
    v16bf a;
    a[0]  = f2bf(a0.x); a[1]  = f2bf(a0.y); a[2]  = f2bf(a0.z); a[3]  = f2bf(a0.w);
    a[4]  = f2bf(a1.x); a[5]  = f2bf(a1.y); a[6]  = f2bf(a1.z); a[7]  = f2bf(a1.w);
    a[8]  = f2bf(a2.x); a[9]  = f2bf(a2.y); a[10] = f2bf(a2.z); a[11] = f2bf(a2.w);
    a[12] = f2bf(a3.x); a[13] = f2bf(a3.y); a[14] = f2bf(a3.z); a[15] = f2bf(a3.w);
    v16bf b;
    const __bf16* vt = &Vt[buf][0];
#pragma unroll
    for (int i = 0; i < 16; ++i) b[i] = vt[(bkb + i + (i >> 3) * 8) * LDSROW_BF + bn];

    acc = __builtin_amdgcn_wmma_f32_16x16x32_bf16(false, a, false, b,
                                                  (short)0, acc, false, false);
    buf ^= 1;
  }

  const int cn  = n0 + (lane & 15);
  const int cmb = t0 + (lane >> 4) * 8;
#pragma unroll
  for (int i = 0; i < 8; ++i)
    ctx[((size_t)bb * T_ + (cmb + i)) * D_ + h * DK_ + cn] = f2bf(acc[i]);
}

// ---------------------------------------------------------------------------
extern "C" void kernel_launch(void* const* d_in, const int* in_sizes, int n_in,
                              void* d_out, int out_size, void* d_ws, size_t ws_size,
                              hipStream_t stream) {
  (void)in_sizes; (void)n_in; (void)out_size; (void)ws_size;

  const float*         x     = (const float*)d_in[0];
  const unsigned char* mask  = (const unsigned char*)d_in[1];
  const float*         pos   = (const float*)d_in[2];
  const float*         W_qkv = (const float*)d_in[3];
  const float*         b_qkv = (const float*)d_in[4];
  const float*         W_pos = (const float*)d_in[5];
  const float*         posu  = (const float*)d_in[6];
  const float*         posv  = (const float*)d_in[7];
  const float*         W_out = (const float*)d_in[8];
  const float*         b_out = (const float*)d_in[9];

  // Output: (context [B,T,D], weights [B,H,T,T]) concatenated, f32
  float* ctx_out = (float*)d_out;
  float* wts     = ctx_out + (size_t)B_ * T_ * D_;

  // Workspace: all bf16 intermediates (~35 MB total)
  __bf16* p = (__bf16*)d_ws;
  __bf16* xb     = p;  p += (size_t)B_ * T_ * D_;    // 2,097,152
  __bf16* posb   = p;  p += (size_t)2048 * D_;       // 1,048,576 (2047 rows used)
  __bf16* Wqkv_b = p;  p += (size_t)D_ * D3_;        //   786,432
  __bf16* Wpos_b = p;  p += (size_t)D_ * D_;         //   262,144
  __bf16* Wout_b = p;  p += (size_t)D_ * D_;         //   262,144
  __bf16* qkvb   = p;  p += (size_t)B_ * T_ * D3_;   // 6,291,456
  __bf16* pprojb = p;  p += (size_t)2048 * D_;       // 1,048,576
  __bf16* qub    = p;  p += (size_t)B_ * T_ * D_;    // 2,097,152
  __bf16* qvb    = p;  p += (size_t)B_ * T_ * D_;    // 2,097,152
  __bf16* ctxb   = p;

  // 0) convert f32 inputs to bf16 residents
  {
    const int n8x  = (B_ * T_ * D_) / 8;
    const int n8p  = (S_ * D_) / 8;
    const int n8wq = (D_ * D3_) / 8;
    const int n8ww = (D_ * D_) / 8;
    f32_to_bf16_kernel<<<(n8x  + 255) / 256, 256, 0, stream>>>(x,     xb,     n8x);
    f32_to_bf16_kernel<<<(n8p  + 255) / 256, 256, 0, stream>>>(pos,   posb,   n8p);
    f32_to_bf16_kernel<<<(n8wq + 255) / 256, 256, 0, stream>>>(W_qkv, Wqkv_b, n8wq);
    f32_to_bf16_kernel<<<(n8ww + 255) / 256, 256, 0, stream>>>(W_pos, Wpos_b, n8ww);
    f32_to_bf16_kernel<<<(n8ww + 255) / 256, 256, 0, stream>>>(W_out, Wout_b, n8ww);
  }

  // 1) qkv = x @ W_qkv + b_qkv   [4096 x 1536] -> bf16
  wmma_gemm<true><<<dim3(D3_ / 16, (B_ * T_) / 64), 128, 0, stream>>>(
      xb, Wqkv_b, b_qkv, qkvb, B_ * T_, D3_, D_);

  // 2) pproj = pos @ W_pos       [2047 x 512] -> bf16
  wmma_gemm<true><<<dim3(D_ / 16, (S_ + 63) / 64), 128, 0, stream>>>(
      posb, Wpos_b, nullptr, pprojb, S_, D_, D_);

  // 3) qu = q + posu, qv = q + posv (bf16)
  quqv_kernel<<<(B_ * T_ * D_) / 8 / 256, 256, 0, stream>>>(
      qkvb, posu, posv, qub, qvb);

  // 4) fused AC + BD(+rel-shift) + mask + scale -> weights region of d_out
  scores_kernel<<<dim3(T_ / 16, T_ / 16, B_ * H_), 32, 0, stream>>>(
      qkvb, qub, qvb, pprojb, mask, wts);

  // 5) row softmax in place
  softmax_kernel<<<dim3(B_ * H_ * T_), 256, 0, stream>>>(wts);

  // 6) context = weights @ v -> bf16
  context_kernel<<<dim3(DK_ / 16, T_ / 64, B_ * H_), 128, 0, stream>>>(wts, qkvb, ctxb);

  // 7) out = context @ W_out + b_out -> f32
  wmma_gemm<false><<<dim3(D_ / 16, (B_ * T_) / 64), 128, 0, stream>>>(
      ctxb, Wout_b, b_out, ctx_out, B_ * T_, D_, D_);
}
